// GraphModel_3281355014343
// MI455X (gfx1250) — matrix operand order repaired and checked by reading, not compiled, not run on hardware
//
#include <hip/hip_runtime.h>
#include <hip/hip_bf16.h>

typedef __attribute__((ext_vector_type(16))) _Float16 v16h;
typedef __attribute__((ext_vector_type(4)))  _Float16 v4h;
typedef __attribute__((ext_vector_type(8)))  float    v8f;
typedef __attribute__((ext_vector_type(4)))  float    v4f;

#define UCNT 50000
#define PCNT 50000
#define NCNT (UCNT + PCNT)
#define ECNT 400000
#define DN   256
#define DE   64
#define NH   8
#define CH   32

// ---------------------------------------------------------------------------
// float atomic max via int/uint ordering trick (init with -inf)
// ---------------------------------------------------------------------------
__device__ __forceinline__ void atomicMaxF(float* addr, float val) {
    if (val >= 0.f) atomicMax((int*)addr, __float_as_int(val));
    else            atomicMin((unsigned int*)addr, __float_as_uint(val));
}

// ---------------------------------------------------------------------------
// Tiled WMMA GEMM: C[M,N] = A[M,K] * B[K,N] + bias (optional), fp32 in/out,
// f16 multiply with f32 accumulate via v_wmma_f32_16x16x32_f16.
// Block tile 64x64 (4 waves, one 16-row strip each), K-step 32, and
// software-pipelined global->register prefetch of the next K tile.
// Requires N % 64 == 0, K % 32 == 0. M row-guarded branch-free on loads;
// epilogue has an unguarded fast path for full tiles.
// ldc = row stride of C (lets us write into column slices / sub-matrices).
// ---------------------------------------------------------------------------
__global__ __launch_bounds__(128) void gemm_f16_wmma(
    const float* __restrict__ A, const float* __restrict__ B,
    const float* __restrict__ bias, float* __restrict__ C,
    int M, int N, int K, int ldc)
{
    __shared__ _Float16 As[64][32];   // M x K tile
    __shared__ _Float16 Bs[64][32];   // N x K tile (B transposed)

    const int tid  = threadIdx.x;
    const int wave = tid >> 5;
    const int lane = tid & 31;
    const int rowBase = blockIdx.x * 64;
    const int colBase = blockIdx.y * 64;
    const int r16 = lane & 15;
    const int kb0 = (lane >> 4) * 8;    // table-order K base (ISA 7.12.2), unused: A fragment uses kh
    const int kh  = (lane >> 4) * 16;   // B fragment K base

    // ---- staging pointers (advance inside K loop; no muls there)
    // A tile: 64 rows x 32 cols; 4 passes of 16 rows; each thread one float4.
    const int aRow  = tid >> 3;          // 0..15  (+16 per pass)
    const int aCol4 = (tid & 7) * 4;     // 0,4,...,28
    const float* aPtr[4];
    bool aOK[4];
    #pragma unroll
    for (int i = 0; i < 4; ++i) {
        int grow = rowBase + aRow + i * 16;
        aOK[i] = (grow < M);
        int crow = aOK[i] ? grow : (M - 1);           // clamp: always in-bounds
        aPtr[i] = A + (size_t)crow * K + aCol4;
    }
    // B tile: 32 k-rows x 64 n-cols; 4 passes of 8 k-rows; one float4 each.
    const int bK  = tid >> 4;            // 0..7 (+8 per pass)
    const int bN4 = (tid & 15) * 4;      // 0..60
    const float* bPtr[4];
    #pragma unroll
    for (int i = 0; i < 4; ++i)
        bPtr[i] = B + (size_t)(bK + i * 8) * N + colBase + bN4;
    const size_t bStep = (size_t)32 * N;

    v8f acc[4];
    #pragma unroll
    for (int nt = 0; nt < 4; ++nt) {
        float bv = bias ? bias[colBase + nt * 16 + r16] : 0.f;
        #pragma unroll
        for (int v = 0; v < 8; ++v) acc[nt][v] = bv;
    }

    // ---- prologue: load first K tile into registers
    const int kIters = K >> 5;
    v4f av[4], bvv[4];
    #pragma unroll
    for (int i = 0; i < 4; ++i) { av[i]  = *(const v4f*)aPtr[i]; aPtr[i] += 32;    }
    #pragma unroll
    for (int i = 0; i < 4; ++i) { bvv[i] = *(const v4f*)bPtr[i]; bPtr[i] += bStep; }

    for (int it = 0; it < kIters; ++it) {
        // ---- convert + store current tile to LDS
        #pragma unroll
        for (int i = 0; i < 4; ++i) {
            v4f a = aOK[i] ? av[i] : (v4f){0.f, 0.f, 0.f, 0.f};   // v_cndmask
            v4h h;
            #pragma unroll
            for (int j = 0; j < 4; ++j) h[j] = (_Float16)a[j];
            *(v4h*)&As[aRow + i * 16][aCol4] = h;                  // ds_store_b64
        }
        #pragma unroll
        for (int i = 0; i < 4; ++i) {
            int kk = bK + i * 8;
            #pragma unroll
            for (int j = 0; j < 4; ++j)
                Bs[bN4 + j][kk] = (_Float16)bvv[i][j];             // transpose scatter
        }
        __syncthreads();

        // ---- prefetch next K tile into registers (overlaps WMMA below)
        if (it + 1 < kIters) {
            #pragma unroll
            for (int i = 0; i < 4; ++i) { av[i]  = *(const v4f*)aPtr[i]; aPtr[i] += 32;    }
            #pragma unroll
            for (int i = 0; i < 4; ++i) { bvv[i] = *(const v4f*)bPtr[i]; bPtr[i] += bStep; }
        }

        // A fragment: lanes 0-15 row M=r16 K=0..15; lanes 16-31 K=16..31
        v16h af;
        #pragma unroll
        for (int j = 0; j < 16; ++j)
            af[j] = As[wave * 16 + r16][kh + j];

        #pragma unroll
        for (int nt = 0; nt < 4; ++nt) {
            // B fragment: lanes 0-15 K=0..15, lanes 16-31 K=16..31, N = r16
            v16h bf;
            #pragma unroll
            for (int j = 0; j < 16; ++j)
                bf[j] = Bs[nt * 16 + r16][kh + j];
            acc[nt] = __builtin_amdgcn_wmma_f32_16x16x32_f16(
                false, af, false, bf, (short)0, acc[nt], false, false);
        }
        __syncthreads();
    }

    // ---- epilogue. D layout: row = (lane>>4)*8 + v, col = lane&15 per tile.
    const int mb = rowBase + wave * 16 + ((lane >> 4) << 3);
    float* cRow = C + (size_t)mb * ldc + colBase + r16;   // one mul per thread
    if (rowBase + 64 <= M) {
        // full tile: unguarded coalesced stores
        #pragma unroll
        for (int nt = 0; nt < 4; ++nt) {
            float* cp = cRow + nt * 16;
            #pragma unroll
            for (int v = 0; v < 8; ++v) { *cp = acc[nt][v]; cp += ldc; }
        }
    } else {
        #pragma unroll
        for (int nt = 0; nt < 4; ++nt) {
            float* cp = cRow + nt * 16;
            #pragma unroll
            for (int v = 0; v < 8; ++v) {
                if (mb + v < M) *cp = acc[nt][v];
                cp += ldc;
            }
        }
    }
}

// ---------------------------------------------------------------------------
// In-place LayerNorm + ReLU, one wave32 per row.
// ---------------------------------------------------------------------------
__global__ __launch_bounds__(256) void ln_relu_kernel(
    float* __restrict__ h, const float* __restrict__ g,
    const float* __restrict__ be, int rows, int width)
{
    int row  = blockIdx.x * 8 + (threadIdx.x >> 5);
    int lane = threadIdx.x & 31;
    if (row >= rows) return;
    float* hp = h + (size_t)row * width;
    float s = 0.f, s2 = 0.f;
    for (int i = lane; i < width; i += 32) { float x = hp[i]; s += x; s2 += x * x; }
    #pragma unroll
    for (int m = 16; m; m >>= 1) { s += __shfl_xor(s, m, 32); s2 += __shfl_xor(s2, m, 32); }
    float inv  = 1.f / (float)width;
    float mu   = s * inv;
    float var  = s2 * inv - mu * mu;
    float rstd = rsqrtf(var + 1e-5f);
    for (int i = lane; i < width; i += 32) {
        float x = (hp[i] - mu) * rstd * g[i] + be[i];
        hp[i] = x > 0.f ? x : 0.f;
    }
}

// ---------------------------------------------------------------------------
// Attention kernels
// ---------------------------------------------------------------------------
__global__ __launch_bounds__(256) void init_softmax_kernel(
    float* __restrict__ amax, float* __restrict__ denom, int n)
{
    int t = blockIdx.x * blockDim.x + threadIdx.x;
    if (t < n) { amax[t] = -__builtin_inff(); denom[t] = 0.f; }
}

__global__ __launch_bounds__(256) void attn_alpha_kernel(
    const float* __restrict__ q, const float* __restrict__ k,
    const float* __restrict__ eproj, const int* __restrict__ ei,
    float* __restrict__ alpha, float* __restrict__ amax)
{
    int t = blockIdx.x * blockDim.x + threadIdx.x;   // E*NH threads
    if (t >= ECNT * NH) return;
    int e = t >> 3, h = t & 7;
    int src = ei[e], dst = ei[ECNT + e];
    const float* qp = q + (size_t)dst * DN + h * CH;
    const float* kp = k + (size_t)src * DN + h * CH;
    const float* ep = eproj + (size_t)e * DN + h * CH;
    float s = 0.f;
    #pragma unroll
    for (int c = 0; c < CH; ++c) s += qp[c] * (kp[c] + ep[c]);
    s *= 0.17677669529663687f;   // 1/sqrt(32)
    alpha[t] = s;
    atomicMaxF(&amax[(size_t)dst * NH + h], s);
}

__global__ __launch_bounds__(256) void attn_exp_kernel(
    const int* __restrict__ ei, float* __restrict__ alpha,
    const float* __restrict__ amax, float* __restrict__ denom)
{
    int t = blockIdx.x * blockDim.x + threadIdx.x;
    if (t >= ECNT * NH) return;
    int e = t >> 3, h = t & 7;
    int dst = ei[ECNT + e];
    float ex = __expf(alpha[t] - amax[(size_t)dst * NH + h]);
    alpha[t] = ex;
    atomicAdd(&denom[(size_t)dst * NH + h], ex);
}

__global__ __launch_bounds__(256) void attn_scatter_kernel(
    const float* __restrict__ v, const float* __restrict__ eproj,
    const int* __restrict__ ei, const float* __restrict__ ex,
    const float* __restrict__ denom, float* __restrict__ accum)
{
    int t = blockIdx.x * blockDim.x + threadIdx.x;   // E*DN threads
    if (t >= ECNT * DN) return;
    int e = t >> 8, j = t & 255, h = j >> 5;
    int src = ei[e], dst = ei[ECNT + e];
    float attn = ex[(size_t)e * NH + h] / (denom[(size_t)dst * NH + h] + 1e-16f);
    float m = (v[(size_t)src * DN + j] + eproj[(size_t)e * DN + j]) * attn;
    atomicAdd(&accum[(size_t)dst * DN + j], m);
}

// ---------------------------------------------------------------------------
// Host launcher
// ---------------------------------------------------------------------------
static inline void launch_gemm(hipStream_t s, const float* A, const float* B,
                               const float* bias, float* C,
                               int M, int N, int K, int ldc)
{
    dim3 grid((M + 63) / 64, N / 64);
    gemm_f16_wmma<<<grid, 128, 0, s>>>(A, B, bias, C, M, N, K, ldc);
}

extern "C" void kernel_launch(void* const* d_in, const int* in_sizes, int n_in,
                              void* d_out, int out_size, void* d_ws, size_t ws_size,
                              hipStream_t stream)
{
    // Input layout (setup_inputs order, dicts flattened in insertion order):
    // 0 prods_only, 1 user_x, 2 info_x, 3 name_x, 4 edge_index, 5 edge_attr,
    // 6..11  user_p  {W1,b1,g,be,W2,b2}
    // 12..17 name_p, 18..23 info_p, 24..29 edge_p
    // 30 proj.W, 31 proj.b
    // 32 Wq 33 bq 34 Wk 35 bk 36 Wv 37 bv 38 We 39 Ws 40 bs
    const float* user_x = (const float*)d_in[1];
    const float* info_x = (const float*)d_in[2];
    const float* name_x = (const float*)d_in[3];
    const int*   eidx   = (const int*)d_in[4];
    const float* eattr  = (const float*)d_in[5];
    const float* const* F = (const float* const*)d_in;
    const float *uW1 = F[6],  *ub1 = F[7],  *ug = F[8],  *ube = F[9],  *uW2 = F[10], *ub2 = F[11];
    const float *nW1 = F[12], *nb1 = F[13], *ng = F[14], *nbe = F[15], *nW2 = F[16], *nb2 = F[17];
    const float *iW1 = F[18], *ib1 = F[19], *ig = F[20], *ibe = F[21], *iW2 = F[22], *ib2 = F[23];
    const float *eW1 = F[24], *eb1 = F[25], *eg = F[26], *ebe = F[27], *eW2 = F[28], *eb2 = F[29];
    const float *pW  = F[30], *pb  = F[31];
    const float *Wq = F[32], *bq = F[33], *Wk = F[34], *bk = F[35], *Wv = F[36], *bv = F[37];
    const float *We = F[38], *Ws = F[39], *bs = F[40];

    // Workspace layout (floats)
    float* ws = (float*)d_ws;
    size_t o = 0;
    float* hbuf    = ws + o; o += (size_t)ECNT * 128;        // MLP hidden (max: edge)
    float* prodcat = ws + o; o += (size_t)PCNT * 512;        // [name_e | info_e]
    float* full    = ws + o; o += (size_t)NCNT * DN;         // concat(user_emb, product_emb)
    float* ebuf    = ws + o; o += (size_t)ECNT * DE;         // edge MLP output
    float* qb      = ws + o; o += (size_t)NCNT * DN;
    float* kb      = ws + o; o += (size_t)NCNT * DN;
    float* vb      = ws + o; o += (size_t)NCNT * DN;
    float* eproj   = ws + o; o += (size_t)ECNT * DN;         // edge_embs @ We
    float* accum   = ws + o; o += (size_t)NCNT * DN;         // skip + messages
    float* alpha   = ws + o; o += (size_t)ECNT * NH;
    float* amax    = ws + o; o += (size_t)NCNT * NH;
    float* denom   = ws + o; o += (size_t)NCNT * NH;

    // ---- product name MLP -> prodcat[:, 0:256]
    launch_gemm(stream, name_x, nW1, nb1, hbuf, PCNT, 512, 256, 512);
    ln_relu_kernel<<<(PCNT + 7) / 8, 256, 0, stream>>>(hbuf, ng, nbe, PCNT, 512);
    launch_gemm(stream, hbuf, nW2, nb2, prodcat, PCNT, 256, 512, 512);
    // ---- product info MLP -> prodcat[:, 256:512]
    launch_gemm(stream, info_x, iW1, ib1, hbuf, PCNT, 512, 256, 512);
    ln_relu_kernel<<<(PCNT + 7) / 8, 256, 0, stream>>>(hbuf, ig, ibe, PCNT, 512);
    launch_gemm(stream, hbuf, iW2, ib2, prodcat + 256, PCNT, 256, 512, 512);
    // ---- proj -> product_emb stored in full[U:]
    launch_gemm(stream, prodcat, pW, pb, full + (size_t)UCNT * DN, PCNT, 256, 512, 256);
    // ---- user MLP -> full[:U]
    launch_gemm(stream, user_x, uW1, ub1, hbuf, UCNT, 512, 256, 512);
    ln_relu_kernel<<<(UCNT + 7) / 8, 256, 0, stream>>>(hbuf, ug, ube, UCNT, 512);
    launch_gemm(stream, hbuf, uW2, ub2, full, UCNT, 256, 512, 256);
    // ---- edge MLP -> ebuf
    launch_gemm(stream, eattr, eW1, eb1, hbuf, ECNT, 128, 64, 128);
    ln_relu_kernel<<<(ECNT + 7) / 8, 256, 0, stream>>>(hbuf, eg, ebe, ECNT, 128);
    launch_gemm(stream, hbuf, eW2, eb2, ebuf, ECNT, 64, 128, 64);
    // ---- projections
    launch_gemm(stream, full, Wq, bq, qb, NCNT, DN, DN, DN);
    launch_gemm(stream, full, Wk, bk, kb, NCNT, DN, DN, DN);
    launch_gemm(stream, full, Wv, bv, vb, NCNT, DN, DN, DN);
    launch_gemm(stream, full, Ws, bs, accum, NCNT, DN, DN, DN);   // skip pre-fills accum
    launch_gemm(stream, ebuf, We, nullptr, eproj, ECNT, DN, DE, DN);
    // ---- attention (segment softmax over dst)
    init_softmax_kernel<<<(NCNT * NH + 255) / 256, 256, 0, stream>>>(amax, denom, NCNT * NH);
    attn_alpha_kernel<<<(ECNT * NH + 255) / 256, 256, 0, stream>>>(qb, kb, eproj, eidx, alpha, amax);
    attn_exp_kernel<<<(ECNT * NH + 255) / 256, 256, 0, stream>>>(eidx, alpha, amax, denom);
    attn_scatter_kernel<<<(ECNT * DN + 255) / 256, 256, 0, stream>>>(vb, eproj, eidx, alpha, denom, accum);
    // ---- outputs: (out[:U], product_emb)
    hipMemcpyAsync(d_out, accum, (size_t)UCNT * DN * sizeof(float),
                   hipMemcpyDeviceToDevice, stream);
    hipMemcpyAsync((float*)d_out + (size_t)UCNT * DN, full + (size_t)UCNT * DN,
                   (size_t)PCNT * DN * sizeof(float), hipMemcpyDeviceToDevice, stream);
    (void)in_sizes; (void)n_in; (void)out_size; (void)ws_size;
}
